// MRNNFixD_6777458393410
// MI455X (gfx1250) — compile-verified, matrix operand order checked by reading, not run on hardware
//
#include <hip/hip_runtime.h>
#include <hip/hip_bf16.h>
#include <stdint.h>

#define T_DIM 512
#define B_DIM 256
#define I_DIM 512
#define H_DIM 1024
#define O_DIM 512
#define K_DIM 32

#define BS_ROWS 64
#define BS_LD   40   // padded row stride (elements): 80B keeps 16B alignment, kills bank conflicts

typedef __attribute__((ext_vector_type(16))) __bf16 v16bf;
typedef __attribute__((ext_vector_type(8)))  __bf16 v8bf;
typedef __attribute__((ext_vector_type(8)))  float  v8f;

// 16-byte packet type expected by the async-to-LDS builtin
typedef int tdm_v4i __attribute__((vector_size(16)));
typedef __attribute__((address_space(1))) tdm_v4i* gptr_v4i;
typedef __attribute__((address_space(3))) tdm_v4i* lptr_v4i;

// ---------------------------------------------------------------------------
// helpers
// ---------------------------------------------------------------------------
__device__ __forceinline__ unsigned short f2bf(float f) {
  union { float f; unsigned u; } v; v.f = f;
  unsigned r = v.u + 0x7FFFu + ((v.u >> 16) & 1u);   // round-to-nearest-even
  return (unsigned short)(r >> 16);
}

// Load one 16x32 bf16 fragment (A-operand layout, ISA 7.12.2) from a
// row-major matrix (global or LDS). base points at (row0, k0); ld = row stride.
__device__ __forceinline__ v16bf load_frag(const unsigned short* base, int ld, int lane) {
  int row = lane & 15;
  int kh  = (lane >> 4) << 3;                     // 0 or 8
  const unsigned short* p = base + row * ld + kh;
  v8bf lo = *reinterpret_cast<const v8bf*>(p);        // K = kh .. kh+7
  v8bf hi = *reinterpret_cast<const v8bf*>(p + 16);   // K = kh+16 .. kh+23
  v16bf a;
#pragma unroll
  for (int e = 0; e < 8; ++e) { a[e] = lo[e]; a[8 + e] = hi[e]; }
  return a;
}

// ---- CDNA5 async global->LDS copy (ASYNCcnt path), 16 bytes per lane -------
__device__ __forceinline__ void async_copy16(const unsigned short* g, unsigned short* l) {
#if __has_builtin(__builtin_amdgcn_global_load_async_to_lds_b128)
  __builtin_amdgcn_global_load_async_to_lds_b128(
      (gptr_v4i)(void*)const_cast<unsigned short*>(g),
      (lptr_v4i)(void*)l, 0, 0);
#else
  unsigned lds_off = (unsigned)(unsigned long long)
      (__attribute__((address_space(3))) unsigned short*)l;
  asm volatile("global_load_async_to_lds_b128 %0, %1, off"
               :: "v"(lds_off), "v"(g) : "memory");
#endif
}

__device__ __forceinline__ void wait_async() {
#if __has_builtin(__builtin_amdgcn_s_wait_asynccnt)
  __builtin_amdgcn_s_wait_asynccnt(0);
#else
  asm volatile("s_wait_asynccnt 0" ::: "memory");
#endif
}

// Stage a 64x32 bf16 weight tile (rows n0..n0+63, cols k0..k0+31) into LDS.
// 256 threads x 16B = 4KB tile; per-lane async transfers.
__device__ __forceinline__ void stage_tile(const unsigned short* W, int ld, int n0, int k0,
                                           unsigned short* buf, int tid) {
  int row   = tid >> 2;            // 0..63
  int chunk = (tid & 3) << 3;      // element offset 0,8,16,24
  const unsigned short* g = W + (size_t)(n0 + row) * ld + k0 + chunk;
  async_copy16(g, buf + row * BS_LD + chunk);
}

// ---------------------------------------------------------------------------
// small prep kernels
// ---------------------------------------------------------------------------
__global__ __launch_bounds__(256) void pis_kernel(const float* __restrict__ b_d,
                                                  float* __restrict__ pis) {
  int i = blockIdx.x * blockDim.x + threadIdx.x;
  if (i >= I_DIM) return;
  float d = 0.5f / (1.0f + __expf(-b_d[i]));          // 0.5*sigmoid
  float cum = 1.0f;
#pragma unroll
  for (int j = 0; j < K_DIM; ++j) {                   // pi_{j+1}
    cum *= ((float)j - d) / ((float)j + 1.0f);
    pis[j * I_DIM + i] = cum;
  }
}

__global__ __launch_bounds__(256) void cvt_bf16_kernel(const float* __restrict__ src,
                                                       unsigned short* __restrict__ dst,
                                                       int n) {
  int i = blockIdx.x * blockDim.x + threadIdx.x;
  if (i < n) dst[i] = f2bf(src[i]);
}

__global__ __launch_bounds__(256) void zero_bf16_kernel(unsigned short* __restrict__ dst,
                                                        int n) {
  int i = blockIdx.x * blockDim.x + threadIdx.x;
  if (i < n) dst[i] = 0;
}

__global__ __launch_bounds__(256) void copy_f32_kernel(const float* __restrict__ src,
                                                       float* __restrict__ dst, int n) {
  int i = blockIdx.x * blockDim.x + threadIdx.x;
  if (i < n) dst[i] = src[i];
}

// ---------------------------------------------------------------------------
// fractional-differencing depthwise conv: U[t,b,i] = x[t,b,i] + sum pi_j x[t-j,b,i]
// ---------------------------------------------------------------------------
__global__ __launch_bounds__(256) void u_kernel(const float* __restrict__ x,
                                                const float* __restrict__ pis,
                                                unsigned short* __restrict__ U) {
  int idx = blockIdx.x * blockDim.x + threadIdx.x;    // over B*I
  if (idx >= B_DIM * I_DIM) return;
  int i = idx % I_DIM;
  float w[K_DIM];
#pragma unroll
  for (int j = 0; j < K_DIM; ++j) w[j] = pis[j * I_DIM + i];

  const size_t stride = (size_t)B_DIM * I_DIM;
  for (int t = 0; t < T_DIM; ++t) {
    float s = x[(size_t)t * stride + idx];
#pragma unroll
    for (int j = 1; j <= K_DIM; ++j) {
      if (t - j >= 0) s += w[j - 1] * x[(size_t)(t - j) * stride + idx];
    }
    U[(size_t)t * stride + idx] = f2bf(s);
  }
}

// ---------------------------------------------------------------------------
// shared inner GEMM phase: acc += A[m0, :Kext] * W[n0..n0+63, :Kext]^T
// B tiles double-buffered through LDS via async copies; A straight from L2.
// ---------------------------------------------------------------------------
__device__ __forceinline__ void gemm_phase(const unsigned short* __restrict__ Abase, int lda,
                                           const unsigned short* __restrict__ W, int ldw,
                                           int Kext, int m0, int n0, int lane, int tid,
                                           unsigned short (*Bs)[BS_ROWS * BS_LD],
                                           v8f acc[4]) {
  stage_tile(W, ldw, n0, 0, Bs[0], tid);
  for (int k0 = 0, it = 0; k0 < Kext; k0 += 32, ++it) {
    wait_async();
    __syncthreads();
    unsigned short* cur = Bs[it & 1];
    if (k0 + 32 < Kext) stage_tile(W, ldw, n0, k0 + 32, Bs[(it & 1) ^ 1], tid);
    v16bf a = load_frag(Abase + (size_t)m0 * lda + k0, lda, lane);
#pragma unroll
    for (int j = 0; j < 4; ++j) {
      v16bf b = load_frag(cur + (16 * j) * BS_LD, BS_LD, lane);
      acc[j] = __builtin_amdgcn_wmma_f32_16x16x32_bf16(false, a, false, b,
                                                       (short)0, acc[j], false, false);
    }
  }
}

// ---------------------------------------------------------------------------
// one recurrence step: h_next = tanh(U_t @ Wx^T + h_prev @ Wh^T + b_h)
// block = 8 waves sharing one 64-wide N strip (weight tile staged in LDS once),
// each wave owns one 16-row M tile. grid = 2 M-blocks x 16 N-groups = 32 blocks.
// ---------------------------------------------------------------------------
__global__ __launch_bounds__(256) void step_kernel(const unsigned short* __restrict__ Ut,    // B x I
                                                   const unsigned short* __restrict__ hprev, // B x H
                                                   const unsigned short* __restrict__ Wx,    // H x I
                                                   const unsigned short* __restrict__ Wh,    // H x H
                                                   const float* __restrict__ bh,             // H
                                                   unsigned short* __restrict__ hnext,       // B x H
                                                   float* __restrict__ hT_out,               // B x H
                                                   int write_f32) {
  __shared__ unsigned short Bs[2][BS_ROWS * BS_LD];

  int tid  = threadIdx.x;
  int lane = tid & 31;
  int warp = tid >> 5;                       // 0..7
  int mb   = blockIdx.x >> 4;                // 0..1
  int ng   = blockIdx.x & 15;                // 0..15
  int m0   = (mb * 8 + warp) << 4;
  int n0   = ng << 6;

  v8f acc[4] = {};

  // phase 1: U_t @ Wx^T   (K = I_DIM)
  gemm_phase(Ut, I_DIM, Wx, I_DIM, I_DIM, m0, n0, lane, tid, Bs, acc);
  __syncthreads();
  // phase 2: h_prev @ Wh^T  (K = H_DIM)
  gemm_phase(hprev, H_DIM, Wh, H_DIM, H_DIM, m0, n0, lane, tid, Bs, acc);

  // epilogue: bias + tanh; C/D layout (VGPR r -> M = r + 8*(lane>=16), N = lane&15)
  int hi = lane >> 4;
  int nc = lane & 15;
#pragma unroll
  for (int j = 0; j < 4; ++j) {
    int n = n0 + 16 * j + nc;
    float bias = bh[n];
#pragma unroll
    for (int r = 0; r < 8; ++r) {
      int m = m0 + r + 8 * hi;
      float h = tanhf(acc[j][r] + bias);
      hnext[(size_t)m * H_DIM + n] = f2bf(h);
      if (write_f32) hT_out[(size_t)m * H_DIM + n] = h;
    }
  }
}

// ---------------------------------------------------------------------------
// final projection: out = h_T @ Wo^T + b_o   (256 x 512, K = 1024)
// ---------------------------------------------------------------------------
__global__ __launch_bounds__(256) void out_kernel(const unsigned short* __restrict__ hT,  // B x H
                                                  const unsigned short* __restrict__ Wo,  // O x H
                                                  const float* __restrict__ bo,           // O
                                                  float* __restrict__ out) {              // B x O
  int lane = threadIdx.x & 31;
  int wave = blockIdx.x * (blockDim.x >> 5) + (threadIdx.x >> 5);
  int m0 = (wave >> 3) << 4;
  int n0 = (wave & 7) << 6;

  v8f acc[4] = {};
  for (int k0 = 0; k0 < H_DIM; k0 += 32) {
    v16bf a = load_frag(hT + (size_t)m0 * H_DIM + k0, H_DIM, lane);
#pragma unroll
    for (int j = 0; j < 4; ++j) {
      v16bf b = load_frag(Wo + (size_t)(n0 + 16 * j) * H_DIM + k0, H_DIM, lane);
      acc[j] = __builtin_amdgcn_wmma_f32_16x16x32_bf16(false, a, false, b,
                                                       (short)0, acc[j], false, false);
    }
  }

  int hi = lane >> 4;
  int nc = lane & 15;
#pragma unroll
  for (int j = 0; j < 4; ++j) {
    int n = n0 + 16 * j + nc;
    float bias = bo[n];
#pragma unroll
    for (int r = 0; r < 8; ++r) {
      int m = m0 + r + 8 * hi;
      out[(size_t)m * O_DIM + n] = acc[j][r] + bias;
    }
  }
}

// ---------------------------------------------------------------------------
// launcher
// ---------------------------------------------------------------------------
extern "C" void kernel_launch(void* const* d_in, const int* in_sizes, int n_in,
                              void* d_out, int out_size, void* d_ws, size_t ws_size,
                              hipStream_t stream) {
  const float* x   = (const float*)d_in[0];   // (T,B,I)
  const float* b_d = (const float*)d_in[1];   // (1,I)
  const float* W_x = (const float*)d_in[2];   // (H,I)
  const float* W_h = (const float*)d_in[3];   // (H,H)
  const float* b_h = (const float*)d_in[4];   // (H)
  const float* W_o = (const float*)d_in[5];   // (O,H)
  const float* b_o = (const float*)d_in[6];   // (O)

  // workspace carve-up
  char* ws = (char*)d_ws;
  size_t off = 0;
  float* pis = (float*)(ws + off);            off += (size_t)K_DIM * I_DIM * sizeof(float);
  unsigned short* Wx_bf = (unsigned short*)(ws + off); off += (size_t)H_DIM * I_DIM * 2;
  unsigned short* Wh_bf = (unsigned short*)(ws + off); off += (size_t)H_DIM * H_DIM * 2;
  unsigned short* Wo_bf = (unsigned short*)(ws + off); off += (size_t)O_DIM * H_DIM * 2;
  unsigned short* hbuf0 = (unsigned short*)(ws + off); off += (size_t)B_DIM * H_DIM * 2;
  unsigned short* hbuf1 = (unsigned short*)(ws + off); off += (size_t)B_DIM * H_DIM * 2;
  unsigned short* Ubf   = (unsigned short*)(ws + off); off += (size_t)T_DIM * B_DIM * I_DIM * 2;

  // outputs: [out_y (B,O) | h_T (B,H) | mem_T (K,B,I)] as f32
  float* out_y = (float*)d_out;
  float* hT_f  = out_y + (size_t)B_DIM * O_DIM;
  float* memT  = hT_f + (size_t)B_DIM * H_DIM;

  // 1) coefficients + weight conversion + h0 = 0
  pis_kernel<<<(I_DIM + 255) / 256, 256, 0, stream>>>(b_d, pis);
  cvt_bf16_kernel<<<(H_DIM * I_DIM + 255) / 256, 256, 0, stream>>>(W_x, Wx_bf, H_DIM * I_DIM);
  cvt_bf16_kernel<<<(H_DIM * H_DIM + 255) / 256, 256, 0, stream>>>(W_h, Wh_bf, H_DIM * H_DIM);
  cvt_bf16_kernel<<<(O_DIM * H_DIM + 255) / 256, 256, 0, stream>>>(W_o, Wo_bf, O_DIM * H_DIM);
  zero_bf16_kernel<<<(B_DIM * H_DIM + 255) / 256, 256, 0, stream>>>(hbuf0, B_DIM * H_DIM);

  // 2) fractionally-differenced inputs for all t (parallel; no recurrence)
  u_kernel<<<(B_DIM * I_DIM + 255) / 256, 256, 0, stream>>>(x, pis, Ubf);

  // 3) mem_T == inputs[T-K : T] flattened (contiguous)
  {
    int n = K_DIM * B_DIM * I_DIM;
    copy_f32_kernel<<<(n + 255) / 256, 256, 0, stream>>>(
        x + (size_t)(T_DIM - K_DIM) * B_DIM * I_DIM, memT, n);
  }

  // 4) sequential recurrence: 512 fused GEMM steps (bf16 WMMA, f32 accum,
  //    weights staged through LDS with async copies)
  unsigned short* hb[2] = {hbuf0, hbuf1};
  for (int t = 0; t < T_DIM; ++t) {
    step_kernel<<<32, 256, 0, stream>>>(Ubf + (size_t)t * B_DIM * I_DIM,
                                        hb[t & 1], Wx_bf, Wh_bf, b_h,
                                        hb[(t + 1) & 1], hT_f,
                                        (t == T_DIM - 1) ? 1 : 0);
  }

  // 5) final output projection from h_T (lives in hbuf0 since T is even)
  out_kernel<<<16, 256, 0, stream>>>(hbuf0, Wo_bf, b_o, out_y);
}
